// IlqlSampler_19524921328039
// MI455X (gfx1250) — matrix operand order repaired and checked by reading, not compiled, not run on hardware
//
#include <hip/hip_runtime.h>
#include <hip/hip_bf16.h>
#include <math.h>

#define FILL_VAL (-1000000000.0f)

typedef __attribute__((ext_vector_type(16))) __bf16 v16bf;
typedef __attribute__((ext_vector_type(8)))  __bf16 bf16x8;
typedef __attribute__((ext_vector_type(8)))  float  v8f;

// ---------------------------------------------------------------------------
// 1) token counts: counts[b, tok] += 1
// ---------------------------------------------------------------------------
__global__ void count_tokens_kernel(const int* __restrict__ tok,
                                    unsigned* __restrict__ counts,
                                    int B, int L, int V) {
  int i = blockIdx.x * blockDim.x + threadIdx.x;
  if (i >= B * L) return;
  int b = i / L;
  int t = tok[i];
  if (t >= 0 && t < V) atomicAdd(&counts[(size_t)b * V + t], 1u);
}

// ---------------------------------------------------------------------------
// 2) logits = H (64x2048) @ E^T (2048x128000) + bias. bf16 WMMA, f32 accum.
//    Block = 256 threads = 8 waves; tile M=64 (ALL batch rows -> embedding is
//    streamed from HBM exactly once), N=128, K chunk = 64.
//    E tile: GLOBAL_LOAD_ASYNC_TO_LDS_B128 (ASYNCcnt DMA, raw f32 in LDS,
//    bf16 conversion at fragment build). H tile: bf16-converted in LDS.
//    Each wave: 1 B-fragment reused across 4 A-fragments -> 8 WMMA / chunk.
// ---------------------------------------------------------------------------
__global__ __launch_bounds__(256)
void gemm_logits_kernel(const float* __restrict__ H, const float* __restrict__ E,
                        const float* __restrict__ EB, float* __restrict__ OUT,
                        int V, int D) {
  __shared__ __bf16 sH[64][72];    // bf16, row stride 144B (16B multiple)
  __shared__ float  sEf[128][68];  // f32,  row stride 272B (16B multiple)

  const int tid  = threadIdx.x;
  const int lane = tid & 31;
  const int wave = tid >> 5;
  const int m0 = blockIdx.y * 64;
  const int v0 = blockIdx.x * 128;
  const int hi  = lane >> 4;   // ISA half-wave lane group
  const int l15 = lane & 15;
  const int nSub = wave * 16;  // this wave's N sub-tile

  v8f acc[4] = {v8f{}, v8f{}, v8f{}, v8f{}};

  for (int k0 = 0; k0 < D; k0 += 64) {
    __syncthreads();  // LDS reuse guard (WAR vs previous chunk's readers)

    // --- stage E chunk 128x64 f32 via async DMA to LDS (8 b128 per thread)
#pragma unroll
    for (int it = 0; it < 8; ++it) {
      int idx = tid + it * 256;
      int row = idx >> 4;        // 0..127
      int c4  = idx & 15;        // float4 index 0..15
      const float* src = E + (size_t)(v0 + row) * D + k0 + c4 * 4;
      unsigned lds_addr = (unsigned)(unsigned long long)(const void*)&sEf[row][c4 * 4];
      unsigned long long gaddr = (unsigned long long)src;
      asm volatile("global_load_async_to_lds_b128 %0, %1, off"
                   :: "v"(lds_addr), "v"(gaddr) : "memory");
      if (k0 + 64 < D) __builtin_prefetch(src + 64, 0, 3);  // global_prefetch_b8
    }
    // --- stage H chunk 64x64 -> bf16 (4 float4 per thread)
#pragma unroll
    for (int it = 0; it < 4; ++it) {
      int idx = tid + it * 256;
      int row = idx >> 4;        // 0..63
      int c4  = idx & 15;
      const float4 hv = *(const float4*)(H + (size_t)(m0 + row) * D + k0 + c4 * 4);
      sH[row][c4 * 4 + 0] = (__bf16)hv.x;
      sH[row][c4 * 4 + 1] = (__bf16)hv.y;
      sH[row][c4 * 4 + 2] = (__bf16)hv.z;
      sH[row][c4 * 4 + 3] = (__bf16)hv.w;
    }
    asm volatile("s_wait_asynccnt 0x0" ::: "memory");  // async DMA complete
    __syncthreads();

    // --- two 16x16x32 bf16 WMMAs per K chunk per M sub-tile
#pragma unroll
    for (int ko = 0; ko < 64; ko += 32) {
      // B 32x16 fragment: lane N=l15 (per half-wave), K = ko + 16*hi + e
      const float* pb = &sEf[nSub + l15][ko + 16 * hi];
      const float4 q0 = *(const float4*)(pb + 0);
      const float4 q1 = *(const float4*)(pb + 4);
      const float4 q2 = *(const float4*)(pb + 8);
      const float4 q3 = *(const float4*)(pb + 12);
      v16bf b;
      b[0]  = (__bf16)q0.x; b[1]  = (__bf16)q0.y; b[2]  = (__bf16)q0.z; b[3]  = (__bf16)q0.w;
      b[4]  = (__bf16)q1.x; b[5]  = (__bf16)q1.y; b[6]  = (__bf16)q1.z; b[7]  = (__bf16)q1.w;
      b[8]  = (__bf16)q2.x; b[9]  = (__bf16)q2.y; b[10] = (__bf16)q2.z; b[11] = (__bf16)q2.w;
      b[12] = (__bf16)q3.x; b[13] = (__bf16)q3.y; b[14] = (__bf16)q3.z; b[15] = (__bf16)q3.w;
#pragma unroll
      for (int ms = 0; ms < 4; ++ms) {
        // A 16x32: lane M=l15; K = ko + 16*hi + e
        const bf16x8 a0 = *(const bf16x8*)&sH[ms * 16 + l15][ko + 16 * hi];
        const bf16x8 a1 = *(const bf16x8*)&sH[ms * 16 + l15][ko + 16 * hi + 8];
        v16bf a;
#pragma unroll
        for (int e = 0; e < 8; ++e) { a[e] = a0[e]; a[e + 8] = a1[e]; }
        acc[ms] = __builtin_amdgcn_wmma_f32_16x16x32_bf16(
            false, a, false, b, (short)0, acc[ms], false, false);
      }
    }
  }

  // C/D layout: VGPR r -> M = r + 8*hi ; N = l15
  const int n = v0 + nSub + l15;
  const float bias = EB[n];
#pragma unroll
  for (int ms = 0; ms < 4; ++ms) {
#pragma unroll
    for (int r = 0; r < 8; ++r) {
      int m = m0 + ms * 16 + r + 8 * hi;
      OUT[(size_t)m * V + n] = acc[ms][r] + bias;
    }
  }
}

// ---------------------------------------------------------------------------
// 3) penalties (+ temperature on logits only); lb penalized into region1
// ---------------------------------------------------------------------------
__global__ void penalize_kernel(float* __restrict__ X, float* __restrict__ LBo,
                                const float* __restrict__ LBi,
                                const unsigned* __restrict__ counts,
                                const float* __restrict__ pres,
                                const float* __restrict__ freq,
                                const float* __restrict__ rep,
                                const float* __restrict__ temp,
                                int V, long N) {
  long i = (long)blockIdx.x * blockDim.x + threadIdx.x;
  if (i >= N) return;
  int b = (int)(i / V);
  float cnt = (float)counts[i];
  bool seen = cnt > 0.0f;
  float rp = rep[b], fq = freq[b], pr = pres[b];

  float x = X[i];
  if (seen) x = (x > 0.0f) ? x / rp : x * rp;
  x = x - fq * cnt - (seen ? pr : 0.0f);
  X[i] = x / temp[b];

  float lb = LBi[i];
  if (seen) lb = (lb > 0.0f) ? lb / rp : lb * rp;
  lb = lb - fq * cnt - (seen ? pr : 0.0f);
  LBo[i] = lb;
}

// ---------------------------------------------------------------------------
// 4) per-row threshold via LDS histogram (replaces the reference's full sort)
//    threshold = max(top_k rank cut, top_p mass cut, m + ln(min_p))
// ---------------------------------------------------------------------------
#define NB 1024
#define HRANGE 32.0f

__global__ __launch_bounds__(256)
void row_threshold_kernel(const float* __restrict__ X, float* __restrict__ stats,
                          const float* __restrict__ topp,
                          const int* __restrict__ topk,
                          const float* __restrict__ minp, int V) {
  __shared__ float red[256];
  __shared__ unsigned hcnt[NB];
  __shared__ float hmass[NB];
  __shared__ float sb[2];

  const int row = blockIdx.x, tid = threadIdx.x;
  const float* x = X + (size_t)row * V;

  // row max
  float lm = -3.0e38f;
  for (int v = tid; v < V; v += 256) lm = fmaxf(lm, x[v]);
  red[tid] = lm;
  __syncthreads();
  for (int s = 128; s > 0; s >>= 1) {
    if (tid < s) red[tid] = fmaxf(red[tid], red[tid + s]);
    __syncthreads();
  }
  if (tid == 0) sb[0] = red[0];
  for (int i = tid; i < NB; i += 256) { hcnt[i] = 0u; hmass[i] = 0.0f; }
  __syncthreads();

  const float m = sb[0];
  const float scale = (float)NB / HRANGE;
  float lz = 0.0f;
  for (int v = tid; v < V; v += 256) {
    float d = m - x[v];
    float p = __expf(-d);
    lz += p;
    if (d < HRANGE) {
      int bin = (int)(d * scale);
      bin = (bin < 0) ? 0 : ((bin > NB - 1) ? NB - 1 : bin);
      atomicAdd(&hcnt[bin], 1u);        // ds_add
      atomicAdd(&hmass[bin], p);        // ds_add_f32
    }
  }
  __syncthreads();
  red[tid] = lz;
  __syncthreads();
  for (int s = 128; s > 0; s >>= 1) {
    if (tid < s) red[tid] += red[tid + s];
    __syncthreads();
  }
  if (tid == 0) {
    float Z = red[0];
    float pthr = topp[row];
    unsigned kthr = (unsigned)topk[row];
    float mp = minp[row];
    const float binW = HRANGE / (float)NB;
    float cumM = 0.0f; unsigned cumC = 0u;
    float tp = -3.0e38f, tk = -3.0e38f;
    bool fp = false, fk = false;
    for (int i = 0; i < NB; ++i) {
      if (!fp && cumM > pthr * Z) { tp = m - i * binW; fp = true; }
      if (!fk && cumC >= kthr)    { tk = m - i * binW; fk = true; }
      cumM += hmass[i];
      cumC += hcnt[i];
    }
    float tm = (mp > 0.0f) ? (m + __logf(mp)) : -3.0e38f;
    stats[row * 2 + 0] = m;
    stats[row * 2 + 1] = fmaxf(fmaxf(tp, tk), tm);
  }
}

// ---------------------------------------------------------------------------
// 5) finalize: masked log-softmax + masked lb, probs = exp, token = argmax
// ---------------------------------------------------------------------------
__global__ __launch_bounds__(256)
void finalize_kernel(float* __restrict__ X, float* __restrict__ LB,
                     float* __restrict__ TOK,
                     const float* __restrict__ statsX,
                     const float* __restrict__ statsL, int V) {
  __shared__ float red[256];
  __shared__ int ridx[256];
  __shared__ float sb[1];

  const int row = blockIdx.x, tid = threadIdx.x;
  const float mx = statsX[row * 2 + 0];
  const float tx = statsX[row * 2 + 1];
  const float tl = statsL[row * 2 + 1];
  float* x  = X  + (size_t)row * V;
  float* lb = LB + (size_t)row * V;

  // lse over masked logits (max element always survives its own thresholds)
  float s = 0.0f;
  for (int v = tid; v < V; v += 256) {
    float xv = x[v];
    if (xv >= tx) s += __expf(xv - mx);
  }
  red[tid] = s;
  __syncthreads();
  for (int st = 128; st > 0; st >>= 1) {
    if (tid < st) red[tid] += red[tid + st];
    __syncthreads();
  }
  if (tid == 0) sb[0] = mx + __logf(red[0]);
  __syncthreads();
  const float lse = sb[0];

  float bestv = -3.0e38f;
  int besti = 0x7fffffff;
  for (int v = tid; v < V; v += 256) {
    float xv = x[v];
    float lv = lb[v];
    float xm  = (xv >= tx) ? xv : FILL_VAL;
    float lbm = (lv >= tl) ? lv : FILL_VAL;
    float lp = (xm - lse) + lbm;
    x[v]  = lp;            // logprobs in-place
    lb[v] = __expf(lp);    // probs in-place
    if (lp > bestv || (lp == bestv && v < besti)) { bestv = lp; besti = v; }
  }
  __syncthreads();
  red[tid] = bestv; ridx[tid] = besti;
  __syncthreads();
  for (int st = 128; st > 0; st >>= 1) {
    if (tid < st) {
      if (red[tid + st] > red[tid] ||
          (red[tid + st] == red[tid] && ridx[tid + st] < ridx[tid])) {
        red[tid] = red[tid + st];
        ridx[tid] = ridx[tid + st];
      }
    }
    __syncthreads();
  }
  if (tid == 0) TOK[row] = (float)ridx[0];
}

// ---------------------------------------------------------------------------
extern "C" void kernel_launch(void* const* d_in, const int* in_sizes, int n_in,
                              void* d_out, int out_size, void* d_ws, size_t ws_size,
                              hipStream_t stream) {
  const float* H    = (const float*)d_in[0];
  const float* E    = (const float*)d_in[1];
  const float* EB   = (const float*)d_in[2];
  const float* LBI  = (const float*)d_in[3];
  const int*   TOKI = (const int*)d_in[4];
  const float* pres = (const float*)d_in[5];
  const float* freq = (const float*)d_in[6];
  const float* rep  = (const float*)d_in[7];
  const float* temp = (const float*)d_in[8];
  const float* topp = (const float*)d_in[9];
  const int*   topk = (const int*)d_in[10];
  const float* minp = (const float*)d_in[11];

  const int V = in_sizes[2];
  const int D = in_sizes[1] / V;
  const int B = in_sizes[0] / D;
  const int L = in_sizes[4] / B;
  const long BV = (long)B * V;

  float* outF   = (float*)d_out;
  float* logits = outF;            // region0: becomes logprobs
  float* lbp    = outF + BV;       // region1: penalized lb -> becomes probs
  float* toks   = outF + 2 * BV;   // region2: tokens (as float)

  unsigned* counts = (unsigned*)d_ws;
  size_t statOff = (((size_t)BV * sizeof(unsigned)) + 255) & ~(size_t)255;
  float* statsX = (float*)((char*)d_ws + statOff);
  float* statsL = statsX + 2 * B;

  hipMemsetAsync(counts, 0, (size_t)BV * sizeof(unsigned), stream);

  {
    int n = B * L;
    count_tokens_kernel<<<(n + 255) / 256, 256, 0, stream>>>(TOKI, counts, B, L, V);
  }
  {
    dim3 grid(V / 128, B / 64);   // one block owns all 64 batch rows: E read once
    gemm_logits_kernel<<<grid, 256, 0, stream>>>(H, E, EB, logits, V, D);
  }
  {
    long N = BV;
    penalize_kernel<<<(unsigned)((N + 255) / 256), 256, 0, stream>>>(
        logits, lbp, LBI, counts, pres, freq, rep, temp, V, N);
  }
  row_threshold_kernel<<<B, 256, 0, stream>>>(logits, statsX, topp, topk, minp, V);
  row_threshold_kernel<<<B, 256, 0, stream>>>(lbp,    statsL, topp, topk, minp, V);
  finalize_kernel<<<B, 256, 0, stream>>>(logits, lbp, toks, statsX, statsL, V);
}
